// BlockSparseMoE_54992761258495
// MI455X (gfx1250) — matrix operand order repaired and checked by reading, not compiled, not run on hardware
//
#include <hip/hip_runtime.h>
#include <hip/hip_bf16.h>

// ---------------- problem sizes (match reference) ----------------
constexpr int kT = 2048;   // tokens
constexpr int kD = 1024;   // d_model
constexpr int kF = 2048;   // ffn hidden per expert
constexpr int kE = 8;      // experts
constexpr int kK = 2;      // top-k
constexpr int kM = 32;     // token tile (2 x 16-row M-fragments)

// ---------------- WMMA types ----------------
typedef __attribute__((ext_vector_type(16))) __bf16 v16bf;
typedef __attribute__((ext_vector_type(8)))  float  v8f;

__device__ inline v8f wmma_bf16(v16bf a, v16bf b, v8f c) {
  // D = A(16x32 bf16) * B(32x16 bf16) + C(16x16 f32)
  return __builtin_amdgcn_wmma_f32_16x16x32_bf16(
      /*neg_a=*/false, a, /*neg_b=*/false, b,
      /*c_mod=*/(short)0, c, /*reuse_a=*/false, /*reuse_b=*/false);
}

__device__ inline unsigned short f2bf(float f) {
  union { __bf16 b; unsigned short s; } u;
  u.b = (__bf16)f;
  return u.s;
}

__device__ inline v8f zero8() {
  v8f z = {0.f, 0.f, 0.f, 0.f, 0.f, 0.f, 0.f, 0.f};
  return z;
}

// A-fragment LDS scatter: 16-bit A matrix 16x32 (M x K), K order matches B:
//   lanes 0-15 : M=lane,    K=0..15
//   lanes 16-31: M=lane-16, K=16..31
// Each lane's 16 bf16 elements are contiguous (32B) -> one v16bf ds load.
__device__ inline void storeA(unsigned short* buf, int m, int k, float val) {
  int lane = m + ((k & 16) ? 16 : 0);
  int elem = k & 15;
  buf[lane * 16 + elem] = f2bf(val);
}

// B-fragment LDS scatter: 16-bit B matrix 32x16 (K x N):
//   lanes 0-15 : N=lane, K=0..15 ; lanes 16-31: N=lane-16, K=16..31
__device__ inline void storeB(unsigned short* buf, int k, int n, float val) {
  int lane = n + ((k & 16) ? 16 : 0);
  int elem = k & 15;
  buf[lane * 16 + elem] = f2bf(val);
}

// Build a B-fragment directly from 16 consecutive fp32 (K-contiguous row data).
__device__ inline v16bf loadB16(const float* p16) {
  union { v16bf v; __bf16 e[16]; } u;
  const float4* p = (const float4*)p16;
#pragma unroll
  for (int q = 0; q < 4; ++q) {
    float4 f = p[q];
    u.e[q * 4 + 0] = (__bf16)f.x;
    u.e[q * 4 + 1] = (__bf16)f.y;
    u.e[q * 4 + 2] = (__bf16)f.z;
    u.e[q * 4 + 3] = (__bf16)f.w;
  }
  return u.v;
}

// ======================= kernel 1: router =======================
__global__ void moe_router(const float* __restrict__ x,
                           const float* __restrict__ gate_w,
                           int* __restrict__ sel, float* __restrict__ wts) {
  int t = blockIdx.x * blockDim.x + threadIdx.x;
  if (t >= kT) return;
  const float* xr = x + (size_t)t * kD;
  float logit[kE];
#pragma unroll
  for (int e = 0; e < kE; ++e) logit[e] = 0.f;
  for (int d = 0; d < kD; ++d) {
    float xv = xr[d];
#pragma unroll
    for (int e = 0; e < kE; ++e) logit[e] = fmaf(xv, gate_w[e * kD + d], logit[e]);
  }
  float mx = logit[0];
#pragma unroll
  for (int e = 1; e < kE; ++e) mx = fmaxf(mx, logit[e]);
  float p[kE], s = 0.f;
#pragma unroll
  for (int e = 0; e < kE; ++e) { p[e] = __expf(logit[e] - mx); s += p[e]; }
  float inv = 1.f / s;
#pragma unroll
  for (int e = 0; e < kE; ++e) p[e] *= inv;
  // stable top-2 (ties -> lowest index, like jax top_k)
  int e0 = 0;
#pragma unroll
  for (int e = 1; e < kE; ++e) if (p[e] > p[e0]) e0 = e;
  int e1 = (e0 == 0) ? 1 : 0;
#pragma unroll
  for (int e = 0; e < kE; ++e) if (e != e0 && p[e] > p[e1]) e1 = e;
  float w0 = p[e0], w1 = p[e1], isw = 1.f / (w0 + w1);
  sel[t * 2 + 0] = e0; sel[t * 2 + 1] = e1;
  wts[t * 2 + 0] = w0 * isw; wts[t * 2 + 1] = w1 * isw;
}

// ================= kernel 2: deterministic compaction =================
__global__ void moe_compact(const int* __restrict__ sel, const float* __restrict__ wts,
                            int* __restrict__ counts, int* __restrict__ tok,
                            float* __restrict__ coef) {
  __shared__ int pc[33];
  int e = blockIdx.x;
  int lane = threadIdx.x;
  constexpr int seg = (kT * kK) / 32;  // entries per lane
  int base = lane * seg;
  int c = 0;
  for (int i = 0; i < seg; ++i) c += (sel[base + i] == e) ? 1 : 0;
  pc[lane + 1] = c;
  __syncthreads();
  if (lane == 0) {
    pc[0] = 0;
    for (int l = 1; l <= 32; ++l) pc[l] += pc[l - 1];
  }
  __syncthreads();
  int off = pc[lane];
  for (int i = 0; i < seg; ++i) {
    int idx = base + i;
    if (sel[idx] == e) {
      tok[(size_t)e * kT + off] = idx >> 1;
      coef[(size_t)e * kT + off] = wts[idx];
      ++off;
    }
  }
  if (lane == 31) counts[e] = pc[32];
}

// ================= kernel 3: fused expert FFN (bf16 WMMA) =================
// Block = (expert e, 32-token tile). 8 waves. M=32 doubles weight reuse vs a
// 16-token tile (weights stream from the 192MB L2; re-read count is the
// bottleneck) and lets every A/B fragment feed 2-4 WMMAs.
__global__ __launch_bounds__(256) void moe_ffn(
    const float* __restrict__ x, const float* __restrict__ w1,
    const float* __restrict__ v1, const float* __restrict__ w2,
    const int* __restrict__ counts, const int* __restrict__ tok,
    const float* __restrict__ coefs, float* __restrict__ out) {
  // x tile: 32 K-chunks x 2 M-fragments, A-frag layout, bf16 -> 64 KB
  __shared__ __attribute__((aligned(32))) unsigned short xa[32 * 2 * 512];
  // h stage: 8 K-chunks x 2 M-fragments -> 16 KB
  __shared__ __attribute__((aligned(32))) unsigned short hst[8 * 2 * 512];
  // per-wave W2 B-fragment staging -> 8 KB
  __shared__ __attribute__((aligned(32))) unsigned short wbuf[8 * 512];
  __shared__ int   s_tok[kM];
  __shared__ float s_coef[kM];

  const int e = blockIdx.x;
  const int tile = blockIdx.y;
  const int count = counts[e];
  if (tile * kM >= count) return;

  const int tid = threadIdx.x;
  const int lane = tid & 31;
  const int wave = tid >> 5;

  if (tid < kM) {
    int slot = tile * kM + tid;
    bool valid = slot < count;
    s_tok[tid] = valid ? tok[(size_t)e * kT + slot] : 0;
    s_coef[tid] = valid ? coefs[(size_t)e * kT + slot] : 0.f;
  }
  __syncthreads();

  // ---- phase 1: gather + convert x tile into A-fragment layout ----
  for (int i = tid; i < kM * (kD / 4); i += 256) {
    int m = i >> 8;            // token row within tile (kD/4 == 256)
    int d4 = i & 255;
    float4 f4 = ((const float4*)(x + (size_t)s_tok[m] * kD))[d4];
    int d = d4 * 4;
    unsigned short* chunk = &xa[((d >> 5) * 2 + (m >> 4)) * 512];
    int mr = m & 15;
    storeA(chunk, mr, (d + 0) & 31, f4.x);
    storeA(chunk, mr, (d + 1) & 31, f4.y);
    storeA(chunk, mr, (d + 2) & 31, f4.z);
    storeA(chunk, mr, (d + 3) & 31, f4.w);
  }
  __syncthreads();

  const int nB = lane & 15;                 // B-fragment column this lane owns
  const int khalf = (lane >= 16) ? 16 : 0;  // B-fragment K half this lane owns
  const int mbase = (lane >= 16) ? 8 : 0;   // C-fragment row base this lane owns

  v8f acc_out[2][8];  // [m-fragment][d-tile]
#pragma unroll
  for (int mi = 0; mi < 2; ++mi)
#pragma unroll
    for (int j = 0; j < 8; ++j) acc_out[mi][j] = zero8();

#pragma unroll 1
  for (int fc = 0; fc < kF / 256; ++fc) {
    // ---- gate/up GEMMs + SiLU: 2 f-tiles per wave, both M-fragments ----
#pragma unroll 1
    for (int p = 0; p < 2; ++p) {
      const int ftl = wave * 2 + p;            // f-tile within this 256 chunk
      const int f = (fc * 16 + ftl) * 16 + nB; // hidden unit for this lane
      const float* w1row = w1 + ((size_t)e * kF + f) * kD + khalf;
      const float* v1row = v1 + ((size_t)e * kF + f) * kD + khalf;
      v8f a1[2] = {zero8(), zero8()};
      v8f a2[2] = {zero8(), zero8()};
#pragma unroll 2
      for (int kc = 0; kc < kD / 32; ++kc) {
        __builtin_prefetch(w1row + kc * 32 + 512, 0, 1);  // global_prefetch_b8
        __builtin_prefetch(v1row + kc * 32 + 512, 0, 1);
        v16bf A0 = *(const v16bf*)&xa[(kc * 2 + 0) * 512 + lane * 16];
        v16bf A1 = *(const v16bf*)&xa[(kc * 2 + 1) * 512 + lane * 16];
        v16bf B1 = loadB16(w1row + kc * 32);
        v16bf B2 = loadB16(v1row + kc * 32);
        a1[0] = wmma_bf16(A0, B1, a1[0]);
        a1[1] = wmma_bf16(A1, B1, a1[1]);
        a2[0] = wmma_bf16(A0, B2, a2[0]);
        a2[1] = wmma_bf16(A1, B2, a2[1]);
      }
      // h = silu(a1) * a2 -> LDS A-fragment layout, both M-fragments
      const int f_local = ftl * 16 + nB;  // [0,256)
      unsigned short* hc0 = &hst[((f_local >> 5) * 2 + 0) * 512];
      unsigned short* hc1 = &hst[((f_local >> 5) * 2 + 1) * 512];
#pragma unroll
      for (int r = 0; r < 8; ++r) {
        float g0 = a1[0][r];
        float g1 = a1[1][r];
        float h0 = (g0 / (1.f + __expf(-g0))) * a2[0][r];
        float h1 = (g1 / (1.f + __expf(-g1))) * a2[1][r];
        storeA(hc0, mbase + r, f_local & 31, h0);
        storeA(hc1, mbase + r, f_local & 31, h1);
      }
    }
    __syncthreads();

    // ---- down GEMM: each wave accumulates its 8 d-tiles over this F chunk ----
    unsigned short* wb = &wbuf[wave * 512];
#pragma unroll 1
    for (int j = 0; j < 8; ++j) {
      const int d0 = (wave * 8 + j) * 16;
#pragma unroll 1
      for (int kcl = 0; kcl < 8; ++kcl) {
        // stage W2 block (rows f = fc*256 + kcl*32 + lane, cols d0..d0+15)
        const int frow = fc * 256 + kcl * 32 + lane;
        const float* wrow = w2 + ((size_t)e * kF + frow) * kD + d0;
        const float4* p4 = (const float4*)wrow;
#pragma unroll
        for (int q = 0; q < 4; ++q) {
          float4 f4 = p4[q];
          storeB(wb, lane, q * 4 + 0, f4.x);
          storeB(wb, lane, q * 4 + 1, f4.y);
          storeB(wb, lane, q * 4 + 2, f4.z);
          storeB(wb, lane, q * 4 + 3, f4.w);
        }
        // DS pipe is in-order within a wave: store->load same buffer is safe.
        v16bf B  = *(const v16bf*)&wb[lane * 16];
        v16bf A0 = *(const v16bf*)&hst[(kcl * 2 + 0) * 512 + lane * 16];
        v16bf A1 = *(const v16bf*)&hst[(kcl * 2 + 1) * 512 + lane * 16];
        acc_out[0][j] = wmma_bf16(A0, B, acc_out[0][j]);
        acc_out[1][j] = wmma_bf16(A1, B, acc_out[1][j]);
      }
    }
    __syncthreads();  // hst reused next fc iteration
  }

  // ---- epilogue: weighted combine into out via f32 atomics ----
  const int nC = lane & 15;
#pragma unroll 1
  for (int mi = 0; mi < 2; ++mi) {
#pragma unroll 1
    for (int j = 0; j < 8; ++j) {
      const int d0 = (wave * 8 + j) * 16;
#pragma unroll
      for (int r = 0; r < 8; ++r) {
        int m = mi * 16 + mbase + r;
        float c = s_coef[m];
        if (c != 0.f)
          atomicAdd(&out[(size_t)s_tok[m] * kD + d0 + nC], c * acc_out[mi][j][r]);
      }
    }
  }
}

// ======================= launch =======================
extern "C" void kernel_launch(void* const* d_in, const int* in_sizes, int n_in,
                              void* d_out, int out_size, void* d_ws, size_t ws_size,
                              hipStream_t stream) {
  (void)in_sizes; (void)n_in; (void)out_size; (void)ws_size;
  const float* x      = (const float*)d_in[0];
  const float* gate_w = (const float*)d_in[1];
  const float* w1     = (const float*)d_in[2];
  const float* v1     = (const float*)d_in[3];
  const float* w2     = (const float*)d_in[4];
  float* out = (float*)d_out;

  // workspace layout (~160 KB)
  int*   sel    = (int*)d_ws;                    // T*K
  float* wts    = (float*)(sel + kT * kK);       // T*K
  int*   counts = (int*)(wts + kT * kK);         // E (+pad)
  int*   tokl   = counts + 16;                   // E*T
  float* coefl  = (float*)(tokl + (size_t)kE * kT);  // E*T

  hipMemsetAsync(d_out, 0, (size_t)kT * kD * sizeof(float), stream);
  moe_router<<<dim3(kT / 256), 256, 0, stream>>>(x, gate_w, sel, wts);
  moe_compact<<<dim3(kE), 32, 0, stream>>>(sel, wts, counts, tokl, coefl);
  moe_ffn<<<dim3(kE, kT / kM), 256, 0, stream>>>(x, w1, v1, w2, counts, tokl, coefl, out);
}
